// SnapshotWeighter_26319559590627
// MI455X (gfx1250) — matrix operand order repaired and checked by reading, not compiled, not run on hardware
//
#include <hip/hip_runtime.h>
#include <hip/hip_bf16.h>

#define D_ 256
#define H_ 4
#define DH_ 64
#define S_ 8192
#define PER_ 32
#define SLOPE_ 0.01f

typedef __attribute__((ext_vector_type(16))) __bf16 v16bf;
typedef __attribute__((ext_vector_type(8)))  float  v8f;

__device__ __forceinline__ __bf16 f2bf(float f) {
  union { float f; unsigned u; } v; v.f = f;
  unsigned r = v.u + 0x7fffu + ((v.u >> 16) & 1u);   // round-to-nearest-even
  unsigned short h = (unsigned short)(r >> 16);
  __bf16 out; __builtin_memcpy(&out, &h, 2);
  return out;
}

__device__ __forceinline__ float leaky(float x) { return x >= 0.f ? x : SLOPE_ * x; }

// A fragment, 16x32 bf16 (ISA 7.12.2): lane holds row M = lane&15;
// K values come as two contiguous 8-element chunks: [k0..k0+7] and [k0+16..k0+23],
// k0 = kb + 8*(lane>=16).
__device__ __forceinline__ v16bf load_fragA(const __bf16* base, int ldr, int mbase, int kb, int lane) {
  const __bf16* p = base + (mbase + (lane & 15)) * ldr + kb + ((lane >> 4) << 3);
  union { v16bf v; uint4 q[2]; } u;
  u.q[0] = *(const uint4*)p;
  u.q[1] = *(const uint4*)(p + 16);
  return u.v;
}

// B fragment, 32x16 bf16: lane holds column N = nb + (lane&15); K in the same chunks as A.
// B[k][n] = W[n][k], stored row-major by n: Wt[n*256 + k].
__device__ __forceinline__ v16bf load_fragB(const __bf16* Wt, int nb, int kb, int lane) {
  const __bf16* p = Wt + (size_t)(nb + (lane & 15)) * 256 + kb + ((lane >> 4) << 3);
  union { v16bf v; uint4 q[2]; } u;
  u.q[0] = *(const uint4*)p;
  u.q[1] = *(const uint4*)(p + 16);
  return u.v;
}

// 32x256 <- (32x256) @ (256x256)^T + bias, bf16 inputs, fp32 accumulate.
// 256 threads = 8 waves; wave w owns N-columns [32w, 32w+32), both 16-row M tiles.
__device__ __forceinline__ void gemm_32x256(const __bf16* __restrict__ A, int lda,
                                            const __bf16* __restrict__ Wt,
                                            const float* __restrict__ bias,
                                            float* out, int ldo) {
  const int tid  = threadIdx.x;
  const int wave = tid >> 5, lane = tid & 31;
  const int nb0  = wave * 32;
  const v8f z = {0.f,0.f,0.f,0.f,0.f,0.f,0.f,0.f};
  v8f acc[4]; acc[0] = z; acc[1] = z; acc[2] = z; acc[3] = z;
#pragma unroll
  for (int kt = 0; kt < 8; ++kt) {
    const int kb = kt * 32;
    v16bf a0 = load_fragA(A, lda, 0,  kb, lane);
    v16bf a1 = load_fragA(A, lda, 16, kb, lane);
    v16bf b0 = load_fragB(Wt, nb0,      kb, lane);
    v16bf b1 = load_fragB(Wt, nb0 + 16, kb, lane);
    acc[0] = __builtin_amdgcn_wmma_f32_16x16x32_bf16(false, a0, false, b0, (short)0, acc[0], false, false);
    acc[1] = __builtin_amdgcn_wmma_f32_16x16x32_bf16(false, a0, false, b1, (short)0, acc[1], false, false);
    acc[2] = __builtin_amdgcn_wmma_f32_16x16x32_bf16(false, a1, false, b0, (short)0, acc[2], false, false);
    acc[3] = __builtin_amdgcn_wmma_f32_16x16x32_bf16(false, a1, false, b1, (short)0, acc[3], false, false);
  }
  // C/D layout: VGPR g, lane l -> row = g + 8*(l>=16), col = l&15
  const int rowo = (lane >> 4) << 3;
  const int col  = lane & 15;
#pragma unroll
  for (int mt = 0; mt < 2; ++mt) {
#pragma unroll
    for (int ntl = 0; ntl < 2; ++ntl) {
      const int n  = nb0 + ntl * 16 + col;
      const float bv = bias[n];
      v8f c = acc[mt * 2 + ntl];
#pragma unroll
      for (int g = 0; g < 8; ++g)
        out[(size_t)(mt * 16 + rowo + g) * ldo + n] = c[g] + bv;
    }
  }
}

// ---------------- prep kernels ----------------
__global__ void cvt_bf16_kernel(const float* __restrict__ src, __bf16* __restrict__ dst, int n) {
  int i = blockIdx.x * blockDim.x + threadIdx.x;
  if (i < n) dst[i] = f2bf(src[i]);
}

__global__ void transpose_cvt_kernel(const float* __restrict__ W, __bf16* __restrict__ Wt) {
  int t = blockIdx.x * blockDim.x + threadIdx.x;   // 65536 threads
  int n = t & 255, k = t >> 8;
  Wt[n * 256 + k] = f2bf(W[n * 256 + k]);
}

// et_tab[s][ic][h] = sum_d leaky(src_type[ic][h*64+d] + snapshot_emb[type(s)][h*64+d]) * attn_t[h][d]
__global__ void et_kernel(const float* __restrict__ snapshot_emb,
                          const float* __restrict__ emb_cite,
                          const float* __restrict__ emb_ref,
                          const float* __restrict__ emb_target,
                          const float* __restrict__ attn_t,
                          const int* __restrict__ types,
                          float* __restrict__ et_tab) {
  int t = blockIdx.x * blockDim.x + threadIdx.x;
  if (t >= S_ * 8) return;
  int h = t & 3, ic = (t >> 2) & 1, s = t >> 3;
  int ty = types[s];
  const float* dst = snapshot_emb + (size_t)ty * 256 + h * 64;
  float sum = 0.f;
#pragma unroll 8
  for (int d = 0; d < 64; ++d) {
    int o = 256 + h * 64 + d;   // embedding row 1 (row 0 is zero / padding)
    float srcv = ic ? (emb_cite[o] + emb_ref[o] + emb_target[o]) : 0.f;
    sum += leaky(srcv + dst[d]) * attn_t[h * 64 + d];
  }
  et_tab[(s * 2 + ic) * 4 + h] = sum;
}

// featD = snapshots @ W_dst^T + b_dst (per 32-row block)
__global__ void __launch_bounds__(256) featD_kernel(const __bf16* __restrict__ snap_bf,
                                                    const __bf16* __restrict__ WdstT,
                                                    const float* __restrict__ b_dst,
                                                    float* __restrict__ featD) {
  size_t off = (size_t)blockIdx.x * 32 * 256;
  gemm_32x256(snap_bf + off, 256, WdstT, b_dst, featD + off, 256);
}

// ---------------- fused per-snapshot kernel ----------------
// 1 block per snapshot (32 contiguous edges): gather -> WMMA GEMM -> logits ->
// segment softmax -> weighted sum, all in LDS.
__global__ void __launch_bounds__(256) fused_snapshot_kernel(
    const float* __restrict__ papers,
    const int*   __restrict__ sel,
    const int*   __restrict__ is_cite,
    const __bf16* __restrict__ WsrcT,
    const float* __restrict__ b_src,
    const float* __restrict__ featD,
    const float* __restrict__ et_tab,
    const float* __restrict__ attn,
    __bf16* __restrict__ sumOut)
{
  __shared__ __align__(16) __bf16 As[32 * 264];   // gathered papers (bf16), padded rows
  __shared__ __align__(16) float  Fs[32 * 264];   // feat_src (fp32)
  __shared__ float fD[256];
  __shared__ float e_l[128];
  __shared__ float a_l[128];
  __shared__ float mh[4], rdh[4];
  __shared__ int   ic_l[32];

  const int s   = blockIdx.x;
  const int tid = threadIdx.x;

  // phase 1: gather 32 paper rows, convert to bf16; load featD row + flags
  {
    const int row  = tid >> 3;            // 0..31
    const int cseg = (tid & 7) * 32;      // 32 floats per thread
    const int pidx = sel[s * 32 + row];
    const float4* src = (const float4*)(papers + (size_t)pidx * 256 + cseg);
    __bf16* dst = As + row * 264 + cseg;
#pragma unroll
    for (int j = 0; j < 8; ++j) {
      float4 v = src[j];
      dst[j * 4 + 0] = f2bf(v.x); dst[j * 4 + 1] = f2bf(v.y);
      dst[j * 4 + 2] = f2bf(v.z); dst[j * 4 + 3] = f2bf(v.w);
    }
    fD[tid] = featD[(size_t)s * 256 + tid];
    if (tid < 32) ic_l[tid] = is_cite[sel[s * 32 + tid]];
  }
  __syncthreads();

  // phase 2: feat_src = As @ W_src^T + b_src  (WMMA)
  gemm_32x256(As, 264, WsrcT, b_src, Fs, 264);
  __syncthreads();

  // phase 3: e[edge][h] = sum_d leaky(feat_src + feat_dst) * attn + et
  if (tid < 128) {
    const int edge = tid >> 2, h = tid & 3;
    const float* frow = Fs + edge * 264 + h * 64;
    const float* fdh  = fD + h * 64;
    const float* at   = attn + h * 64;
    float sum = 0.f;
#pragma unroll 8
    for (int d = 0; d < 64; ++d)
      sum += leaky(frow[d] + fdh[d]) * at[d];
    sum += et_tab[(s * 2 + ic_l[edge]) * 4 + h];
    e_l[tid] = sum;
  }
  __syncthreads();

  // phase 4: per-head softmax over the 32 edges of this segment
  if (tid < 4) {
    float m = -3.402823466e38f;
    for (int e = 0; e < 32; ++e) m = fmaxf(m, e_l[e * 4 + tid]);
    float den = 0.f;
    for (int e = 0; e < 32; ++e) den += __expf(e_l[e * 4 + tid] - m);
    mh[tid] = m; rdh[tid] = 1.f / den;
  }
  __syncthreads();
  if (tid < 128) {
    const int h = tid & 3;
    a_l[tid] = __expf(e_l[tid] - mh[h]) * rdh[h];
  }
  __syncthreads();

  // phase 5: sum_out[d] = sum_e a[e][d/64] * feat_src[e][d]  (store bf16 for final GEMM)
  {
    const int h = tid >> 6;
    float acc = 0.f;
#pragma unroll 8
    for (int e = 0; e < 32; ++e)
      acc += a_l[e * 4 + h] * Fs[e * 264 + tid];
    sumOut[(size_t)s * 256 + tid] = f2bf(acc);
  }
}

// final: out = sumOut @ W_out^T + b_out
__global__ void __launch_bounds__(256) out_gemm_kernel(const __bf16* __restrict__ sumOut,
                                                       const __bf16* __restrict__ WoutT,
                                                       const float* __restrict__ b_out,
                                                       float* __restrict__ out) {
  size_t off = (size_t)blockIdx.x * 32 * 256;
  gemm_32x256(sumOut + off, 256, WoutT, b_out, out + off, 256);
}

extern "C" void kernel_launch(void* const* d_in, const int* in_sizes, int n_in,
                              void* d_out, int out_size, void* d_ws, size_t ws_size,
                              hipStream_t stream) {
  const float* papers       = (const float*)d_in[0];
  const float* snapshots    = (const float*)d_in[1];
  const float* W_src        = (const float*)d_in[2];
  const float* b_src        = (const float*)d_in[3];
  const float* W_dst        = (const float*)d_in[4];
  const float* b_dst        = (const float*)d_in[5];
  const float* W_out        = (const float*)d_in[6];
  const float* b_out        = (const float*)d_in[7];
  const float* emb_cite     = (const float*)d_in[8];
  const float* emb_ref      = (const float*)d_in[9];
  const float* emb_target   = (const float*)d_in[10];
  const float* snapshot_emb = (const float*)d_in[11];
  const float* attn         = (const float*)d_in[12];
  const float* attn_t       = (const float*)d_in[13];
  const int*   cur_types    = (const int*)d_in[14];
  const int*   is_cite      = (const int*)d_in[15];
  const int*   sel          = (const int*)d_in[16];
  // d_in[17] = index: contiguous repeat(arange(S), 32) by construction -> implicit

  char* ws = (char*)d_ws;
  __bf16* WsrcT   = (__bf16*)(ws + 0);          // 128 KB  (W^T, bf16)
  __bf16* WdstT   = (__bf16*)(ws + 131072);     // 128 KB
  __bf16* WoutT   = (__bf16*)(ws + 262144);     // 128 KB
  __bf16* snap_bf = (__bf16*)(ws + 393216);     // 4 MB
  float*  featD   = (float*) (ws + 4587520);    // 8 MB
  float*  et_tab  = (float*) (ws + 12976128);   // 256 KB
  __bf16* sumOut  = (__bf16*)(ws + 13238272);   // 4 MB   (total ~16.6 MB)

  transpose_cvt_kernel<<<256, 256, 0, stream>>>(W_src, WsrcT);
  transpose_cvt_kernel<<<256, 256, 0, stream>>>(W_dst, WdstT);
  transpose_cvt_kernel<<<256, 256, 0, stream>>>(W_out, WoutT);
  cvt_bf16_kernel<<<(S_ * 256 + 255) / 256, 256, 0, stream>>>(snapshots, snap_bf, S_ * 256);
  featD_kernel<<<S_ / 32, 256, 0, stream>>>(snap_bf, WdstT, b_dst, featD);
  et_kernel<<<(S_ * 8 + 255) / 256, 256, 0, stream>>>(snapshot_emb, emb_cite, emb_ref,
                                                      emb_target, attn_t, cur_types, et_tab);
  fused_snapshot_kernel<<<S_, 256, 0, stream>>>(papers, sel, is_cite, WsrcT, b_src,
                                                featD, et_tab, attn, sumOut);
  out_gemm_kernel<<<S_ / 32, 256, 0, stream>>>(sumOut, WoutT, b_out, (float*)d_out);
}
